// StockModel_10754598109658
// MI455X (gfx1250) — matrix operand order repaired and checked by reading, not compiled, not run on hardware
//
#include <hip/hip_runtime.h>
#include <hip/hip_bf16.h>
#include <cstddef>

// ---------------------------------------------------------------------------
// StockModel fused pipeline for MI455X (gfx1250, wave32, WMMA).
//   K0: prep — W1 (200x800 f32) -> zero-padded f16 (208x800); Wt -> f16
//   K1: price LSTM  (T,N,1) -> p_seq (T,N,32)              [1 wave / stock]
//   K2: per-(t,n) fused gather + VertexConv + EdgeConv     [256 thr / (t,n)]
//       - mult einsum  : 64x v_wmma_f32_16x16x32_f16 (M=16 edges per row i)
//       - trans einsum : 4x WMMA per edge, wk-pool folded into D epilogue
//       - EdgeConv GEMM: 16x200x800 via 325x WMMA
//   K3: temporal LSTM over hg + dense heads -> out (N,2)   [1 wave / stock]
// All matrix FLOPs (~14.5 of ~15 GFLOP) run on the WMMA pipes.
// ---------------------------------------------------------------------------

#define TT   4
#define NN   500
#define EE   16
#define KK   32
#define HH   32
#define DD   768
#define ZDIM 800     // HH + DD
#define EHID 200
#define NPAD 208     // EHID padded to multiple of 16 for WMMA N-tiles

typedef __attribute__((ext_vector_type(16))) _Float16 v16h;
typedef __attribute__((ext_vector_type(8)))  float    v8f;

__device__ __forceinline__ float sigm(float x) { return 1.0f / (1.0f + __expf(-x)); }

// A-frag 16x32, this code's contiguous k order (matches the B frags): lanes<16 hold K{0..15}, lanes>=16 hold
// K{16..31} (off=2*aKoff).  `row` points at the 32-K row (stride 1).
__device__ __forceinline__ v16h load_a_frag(const _Float16* __restrict__ row, int aKoff) {
    union { v16h v; _Float16 h[16]; } A;
    #pragma unroll
    for (int p = 0; p < 8; ++p) { A.h[p] = row[2 * aKoff + p]; A.h[8 + p] = row[2 * aKoff + 8 + p]; }
    return A.v;
}
// B-frag, 32x16: lane = column, halfs = 16 contiguous K values (pre-offset ptr).
__device__ __forceinline__ v16h load_b_frag_contig(const _Float16* __restrict__ p) {
    union { v16h v; _Float16 h[16]; } B;
    #pragma unroll
    for (int i = 0; i < 16; ++i) B.h[i] = p[i];
    return B.v;
}
// B-frag from row-major [K][N] LDS tile: element (bKoff+p, col).
__device__ __forceinline__ v16h load_b_frag_strided(const _Float16* __restrict__ base,
                                                    int stride, int col, int bKoff) {
    union { v16h v; _Float16 h[16]; } B;
    #pragma unroll
    for (int p = 0; p < 16; ++p) B.h[p] = base[(bKoff + p) * stride + col];
    return B.v;
}

// ---------------------------------------------------------------------------
// K0: convert W1 (zero-padded to NPAD rows) and Wt to f16
// ---------------------------------------------------------------------------
__global__ void prep_kernel(const float* __restrict__ W1, const float* __restrict__ Wt,
                            _Float16* __restrict__ W1h, _Float16* __restrict__ Wth) {
    int idx = blockIdx.x * 256 + threadIdx.x;
    if (idx < NPAD * ZDIM) {
        int row = idx / ZDIM;
        W1h[idx] = (row < EHID) ? (_Float16)W1[idx] : (_Float16)0.0f;
    }
    int j = idx - NPAD * ZDIM;
    if (j >= 0 && j < KK * KK * HH) Wth[j] = (_Float16)Wt[j];
}

// ---------------------------------------------------------------------------
// K1: price LSTM, one wave32 per stock; lane j owns h[j], c[j].
// ---------------------------------------------------------------------------
__global__ __launch_bounds__(32) void lstm1_kernel(
    const float* __restrict__ prices,
    const float* __restrict__ Wih, const float* __restrict__ Whh,
    const float* __restrict__ bih, const float* __restrict__ bhh,
    float* __restrict__ p_seq)
{
    __shared__ float h_sh[HH];
    const int n = blockIdx.x;
    const int j = threadIdx.x;
    float c = 0.0f;
    h_sh[j] = 0.0f;
    __syncthreads();
    for (int t = 0; t < TT; ++t) {
        float x = prices[t * NN + n];
        float za[4];
        #pragma unroll
        for (int g = 0; g < 4; ++g) {
            int r = g * HH + j;
            float acc = x * Wih[r] + bih[r] + bhh[r];
            const float* wr = Whh + r * HH;
            #pragma unroll
            for (int k = 0; k < HH; ++k) acc += h_sh[k] * wr[k];
            za[g] = acc;
        }
        __syncthreads();
        float ig = sigm(za[0]), fg = sigm(za[1]);
        float gg = tanhf(za[2]), og = sigm(za[3]);
        c = fg * c + ig * gg;
        float h = og * tanhf(c);
        h_sh[j] = h;
        p_seq[((size_t)t * NN + n) * HH + j] = h;
        __syncthreads();
    }
}

// ---------------------------------------------------------------------------
// K2: fused hypergraph block per (t,n). 8 waves, 64 KB phase-aliased LDS.
// ---------------------------------------------------------------------------
__global__ __launch_bounds__(256) void edge_kernel(
    const float* __restrict__ p_seq,      // (T,N,H)
    const float* __restrict__ node_embs,  // (T,N,D)
    const int*   __restrict__ adj,        // (T,N,E)
    const int*   __restrict__ mem,        // (T,N,K)
    const _Float16* __restrict__ Wth,     // (K,K,H) f16
    const float* __restrict__ bt,         // (K,K)
    const float* __restrict__ wk,         // (K)
    const float* __restrict__ bk,         // (1)
    const _Float16* __restrict__ W1h,     // (NPAD, ZDIM) f16, zero-padded
    const float* __restrict__ b1,         // (EHID)
    const float* __restrict__ W2,         // (EHID)
    const float* __restrict__ b2,         // (1)
    float* __restrict__ hg)               // (T,N,ZDIM)
{
    // 64 KB LDS, aliased by phase:
    //   phases 1-4: regh [16][32][32] f16 | multh [16][32][32] f16
    //   phases 5-7: zh   [16][800]    f16 | s_sh[16], att_sh[16] f32
    __shared__ _Float16 smem[32768];
    _Float16* regh   = smem;                       // 16384 halfs
    _Float16* multh  = smem + 16384;               // 16384 halfs
    _Float16* zh     = smem;                       // 12800 halfs
    float*    s_sh   = (float*)(smem + 16384);     // 16 f32
    float*    att_sh = s_sh + EE;                  // 16 f32

    const int tn   = blockIdx.x;                   // 0 .. T*N-1
    const int t    = tn / NN, n = tn % NN;
    const int tid  = threadIdx.x;
    const int w    = tid >> 5;                     // wave id 0..7
    const int lane = tid & 31;
    const int hi   = lane >> 4;                    // half-wave 0/1
    const int aKoff = hi ? 8 : 0;
    const int bKoff = hi ? 16 : 0;
    const int Moff  = hi ? 8 : 0;
    const int l15   = lane & 15;

    // ---- Phase 1: gather region rows for all 16 edges into f16 LDS ----
    #pragma unroll
    for (int ee = 0; ee < 2; ++ee) {
        const int e = w + 8 * ee;
        const int a = adj[(t * NN + n) * EE + e];
        const int m = mem[(t * NN + a) * KK + lane];
        const float* pr = p_seq + (size_t)(t * NN + m) * HH;
        _Float16* dst = regh + (e * KK + lane) * KK;
        #pragma unroll
        for (int d = 0; d < HH; ++d) dst[d] = (_Float16)pr[d];
    }
    __syncthreads();

    // ---- Phase 2: mult[e,i,j] = region[e,i,:].Wt[i,j,:] + bt[i,j] (WMMA) ----
    // For fixed i: A(e,d) = regh[e][i][d] (M = 16 edges), B(d,j) = Wth[i][j][d].
    #pragma unroll
    for (int ii = 0; ii < 4; ++ii) {
        const int i = w + 8 * ii;                  // 32 i-rows over 8 waves
        v16h Af = load_a_frag(regh + (l15 * KK + i) * KK, aKoff);
        #pragma unroll
        for (int jt = 0; jt < 2; ++jt) {
            const int j = jt * 16 + l15;
            v16h Bf = load_b_frag_contig(Wth + ((size_t)i * KK + j) * KK + bKoff);
            v8f acc = {};
            acc = __builtin_amdgcn_wmma_f32_16x16x32_f16(
                false, Af, false, Bf, (short)0, acc, false, false);
            #pragma unroll
            for (int r = 0; r < 8; ++r) {
                int e = r + Moff;
                multh[(e * KK + i) * KK + j] = (_Float16)(acc[r] + bt[i * KK + j]);
            }
        }
    }
    __syncthreads();

    // ---- Phase 3: softmax over j for each (e,i) row, in place ----
    for (int row = tid; row < EE * KK; row += 256) {
        _Float16* mr = multh + row * KK;
        float v[KK];
        float mx = -1e30f;
        #pragma unroll
        for (int j = 0; j < KK; ++j) { v[j] = (float)mr[j]; mx = fmaxf(mx, v[j]); }
        float se = 0.0f;
        #pragma unroll
        for (int j = 0; j < KK; ++j) { v[j] = __expf(v[j] - mx); se += v[j]; }
        float inv = 1.0f / se;
        #pragma unroll
        for (int j = 0; j < KK; ++j) mr[j] = (_Float16)(v[j] * inv);
    }
    __syncthreads();

    // ---- Phase 4: trans[e] = mult[e] @ region[e] (WMMA), fold wk pooling ----
    // vc[e][d] = sum_i wk[i] * trans[e][i][d]; per lane: partial over its 16
    // i-values (8 per M-tile), combined with the other half-wave via shfl_xor.
    float vc[2][2];
    #pragma unroll
    for (int ee = 0; ee < 2; ++ee) {
        const int e = w + 8 * ee;
        #pragma unroll
        for (int nt = 0; nt < 2; ++nt) {
            float part = 0.0f;
            #pragma unroll
            for (int mt = 0; mt < 2; ++mt) {
                v16h Af = load_a_frag(multh + (e * KK + mt * 16 + l15) * KK, aKoff);
                v16h Bf = load_b_frag_strided(regh + e * KK * KK, KK, nt * 16 + l15, bKoff);
                v8f acc = {};
                acc = __builtin_amdgcn_wmma_f32_16x16x32_f16(
                    false, Af, false, Bf, (short)0, acc, false, false);
                #pragma unroll
                for (int r = 0; r < 8; ++r) part += wk[mt * 16 + r + Moff] * acc[r];
            }
            part += __shfl_xor(part, 16);
            vc[ee][nt] = part;
        }
    }
    __syncthreads();   // all regh/multh reads done; LDS block is re-purposed now

    // ---- Phase 5: build z tile zh = [vc | edge_emb], init scores ----
    const float bkv = bk[0];
    #pragma unroll
    for (int ee = 0; ee < 2; ++ee) {
        const int e = w + 8 * ee;
        if (lane < 16) {
            #pragma unroll
            for (int nt = 0; nt < 2; ++nt)
                zh[e * ZDIM + nt * 16 + l15] = (_Float16)(vc[ee][nt] + bkv);
        }
        const int a = adj[(t * NN + n) * EE + e];
        const float* emb = node_embs + (size_t)(t * NN + a) * DD;
        for (int q = lane; q < DD; q += 32) zh[e * ZDIM + HH + q] = (_Float16)emb[q];
    }
    if (tid < EE) s_sh[tid] = b2[0];
    __syncthreads();

    // ---- Phase 6: y = z(16x800) @ W1h^T(800x208) via WMMA, score epilogue ----
    for (int nt = w; nt < NPAD / 16; nt += 8) {    // 13 N-tiles over 8 waves
        v8f acc = {};
        const _Float16* wbase = W1h + (size_t)(nt * 16 + l15) * ZDIM + bKoff;
        for (int kt = 0; kt < ZDIM / 32; ++kt) {   // 25 K-steps
            v16h Af = load_a_frag(zh + l15 * ZDIM + kt * 32, aKoff);
            v16h Bf = load_b_frag_contig(wbase + kt * 32);
            if (kt + 1 < ZDIM / 32) __builtin_prefetch(wbase + kt * 32 + 32, 0, 1);
            acc = __builtin_amdgcn_wmma_f32_16x16x32_f16(
                false, Af, false, Bf, (short)0, acc, false, false);
        }
        const int ncol = nt * 16 + l15;
        if (ncol < EHID) {
            const float b1v = b1[ncol], w2v = W2[ncol];
            #pragma unroll
            for (int r = 0; r < 8; ++r) {
                float y = fmaxf(acc[r] + b1v, 0.0f);
                atomicAdd(&s_sh[r + Moff], y * w2v);
            }
        }
    }
    __syncthreads();

    // ---- Phase 7: softmax over edges, attention-weighted sum -> hg ----
    if (tid == 0) {
        float mx = s_sh[0];
        for (int e2 = 1; e2 < EE; ++e2) mx = fmaxf(mx, s_sh[e2]);
        float se = 0.0f;
        for (int e2 = 0; e2 < EE; ++e2) { float v = __expf(s_sh[e2] - mx); att_sh[e2] = v; se += v; }
        float inv = 1.0f / se;
        for (int e2 = 0; e2 < EE; ++e2) att_sh[e2] *= inv;
    }
    __syncthreads();
    float* hgrow = hg + (size_t)tn * ZDIM;
    for (int j = tid; j < ZDIM; j += 256) {
        float acc = 0.0f;
        #pragma unroll
        for (int e2 = 0; e2 < EE; ++e2) acc += att_sh[e2] * (float)zh[e2 * ZDIM + j];
        hgrow[j] = acc;
    }
}

// ---------------------------------------------------------------------------
// K3: temporal LSTM over hg + heads, one wave32 per stock.
// ---------------------------------------------------------------------------
__global__ __launch_bounds__(32) void lstm2_head_kernel(
    const float* __restrict__ hg,
    const float* __restrict__ Wih, const float* __restrict__ Whh,
    const float* __restrict__ bih, const float* __restrict__ bhh,
    const float* __restrict__ Wf1, const float* __restrict__ bf1,
    const float* __restrict__ Wf2, const float* __restrict__ bf2,
    float* __restrict__ out)
{
    __shared__ float h_sh[HH];
    __shared__ float f1_sh[2 * HH];
    const int n = blockIdx.x;
    const int j = threadIdx.x;
    float c = 0.0f;
    h_sh[j] = 0.0f;
    __syncthreads();

    const int r0 = j, r1 = HH + j, r2 = 2 * HH + j, r3 = 3 * HH + j;
    for (int t = 0; t < TT; ++t) {
        const float* xt = hg + ((size_t)t * NN + n) * ZDIM;
        float z0 = bih[r0] + bhh[r0], z1 = bih[r1] + bhh[r1];
        float z2 = bih[r2] + bhh[r2], z3 = bih[r3] + bhh[r3];
        for (int q = 0; q < ZDIM; ++q) {
            float xq = xt[q];                      // broadcast across lanes
            z0 += xq * Wih[(size_t)r0 * ZDIM + q];
            z1 += xq * Wih[(size_t)r1 * ZDIM + q];
            z2 += xq * Wih[(size_t)r2 * ZDIM + q];
            z3 += xq * Wih[(size_t)r3 * ZDIM + q];
        }
        #pragma unroll
        for (int k = 0; k < HH; ++k) {
            float hk = h_sh[k];
            z0 += hk * Whh[r0 * HH + k];
            z1 += hk * Whh[r1 * HH + k];
            z2 += hk * Whh[r2 * HH + k];
            z3 += hk * Whh[r3 * HH + k];
        }
        __syncthreads();
        float ig = sigm(z0), fg = sigm(z1), gg = tanhf(z2), og = sigm(z3);
        c = fg * c + ig * gg;
        h_sh[j] = og * tanhf(c);
        __syncthreads();
    }

    #pragma unroll
    for (int g = 0; g < 2; ++g) {
        int r = g * HH + j;
        float acc = bf1[r];
        #pragma unroll
        for (int k = 0; k < HH; ++k) acc += h_sh[k] * Wf1[r * HH + k];
        f1_sh[r] = acc;
    }
    __syncthreads();
    if (j < 2) {
        float acc = bf2[j];
        #pragma unroll
        for (int q = 0; q < 2 * HH; ++q) acc += f1_sh[q] * Wf2[j * 2 * HH + q];
        out[n * 2 + j] = acc;
    }
}

// ---------------------------------------------------------------------------
// Launch. Workspace layout (~7.5 MB, all f16/f32 scratch):
//   [0,       256000)   p_seq (T,N,H)    f32
//   [262144, +332800)   W1h   (NPAD,ZDIM) f16
//   [598016, +65536)    Wth   (K,K,H)    f16
//   [1048576,+6.4MB)    hg    (T,N,ZDIM) f32
// ---------------------------------------------------------------------------
extern "C" void kernel_launch(void* const* d_in, const int* in_sizes, int n_in,
                              void* d_out, int out_size, void* d_ws, size_t ws_size,
                              hipStream_t stream) {
    (void)in_sizes; (void)n_in; (void)out_size; (void)ws_size;
    const float* prices    = (const float*)d_in[0];
    const float* node_embs = (const float*)d_in[1];
    const int*   adj       = (const int*)d_in[2];
    const int*   mem       = (const int*)d_in[3];
    const float* W_ih1 = (const float*)d_in[4];
    const float* W_hh1 = (const float*)d_in[5];
    const float* b_ih1 = (const float*)d_in[6];
    const float* b_hh1 = (const float*)d_in[7];
    const float* Wt    = (const float*)d_in[8];
    const float* bt    = (const float*)d_in[9];
    const float* wk    = (const float*)d_in[10];
    const float* bk    = (const float*)d_in[11];
    const float* W1    = (const float*)d_in[12];
    const float* b1    = (const float*)d_in[13];
    const float* W2    = (const float*)d_in[14];
    const float* b2    = (const float*)d_in[15];
    const float* W_ih2 = (const float*)d_in[16];
    const float* W_hh2 = (const float*)d_in[17];
    const float* b_ih2 = (const float*)d_in[18];
    const float* b_hh2 = (const float*)d_in[19];
    const float* Wf1   = (const float*)d_in[20];
    const float* bf1   = (const float*)d_in[21];
    const float* Wf2   = (const float*)d_in[22];
    const float* bf2   = (const float*)d_in[23];
    float* out = (float*)d_out;

    char* ws = (char*)d_ws;
    float*    p_seq = (float*)(ws);
    _Float16* W1h   = (_Float16*)(ws + 262144);
    _Float16* Wth   = (_Float16*)(ws + 598016);
    float*    hg    = (float*)(ws + 1048576);

    const int prep_elems = NPAD * ZDIM + KK * KK * HH;
    prep_kernel<<<(prep_elems + 255) / 256, 256, 0, stream>>>(W1, Wt, W1h, Wth);
    lstm1_kernel<<<NN, 32, 0, stream>>>(prices, W_ih1, W_hh1, b_ih1, b_hh1, p_seq);
    edge_kernel<<<TT * NN, 256, 0, stream>>>(p_seq, node_embs, adj, mem,
                                             Wth, bt, wk, bk,
                                             W1h, b1, W2, b2, hg);
    lstm2_head_kernel<<<NN, 32, 0, stream>>>(hg, W_ih2, W_hh2, b_ih2, b_hh2,
                                             Wf1, bf1, Wf2, bf2, out);
}
